// Backbone_59605556133956
// MI455X (gfx1250) — compile-verified, matrix operand order checked by reading, not run on hardware
//
#include <hip/hip_runtime.h>
#include <cstdint>

// Point Transformer backbone for MI455X (gfx1250, wave32).
// Heavy matmuls -> v_wmma_f32_16x16x32_f16 (f16 in, f32 acc).
// Weights stored in WMMA-fragment-packed layout so every B operand is one
// contiguous 32B load per lane (2x global_load_b128), coalesced per wave.
// Per-point attention fully fused in LDS: K=16 neighbors == WMMA M=16.

typedef __attribute__((ext_vector_type(16))) _Float16 v16h;
typedef __attribute__((ext_vector_type(8)))  float    v8f;

#define KNN 16
#define DM  256

// ---------------- conversions ----------------
__global__ void k_cvt_f16(const float* __restrict__ a, _Float16* __restrict__ o, int n) {
  int i = blockIdx.x * blockDim.x + threadIdx.x;
  if (i < n) o[i] = (_Float16)a[i];
}

// Pack W[Cin,Cout] (row-major f32) into WMMA B-fragment order:
// Wp[((ct*(Cin/32) + ks)*32 + lane)*16 + j] = W[k, ct*16 + (lane&15)]
// with kb=(lane>>4)*8, k = ks*32 + kb + (j<8 ? j : 8+j).
__global__ void k_pack_w(const float* __restrict__ W, _Float16* __restrict__ Wp,
                         int Cin, int Cout) {
  int g = blockIdx.x * blockDim.x + threadIdx.x;
  if (g >= Cin * Cout) return;
  int j    = g & 15;
  int lane = (g >> 4) & 31;
  int rest = g >> 9;
  int nks  = Cin >> 5;
  int ks   = rest % nks;
  int ct   = rest / nks;
  int n    = ct * 16 + (lane & 15);
  int kb   = (lane >> 4) << 3;
  int k    = ks * 32 + kb + (j < 8 ? j : 8 + j);
  Wp[g] = (_Float16)W[(size_t)k * Cout + n];
}

// ---------------- input stem ----------------
__global__ void k_xyz_extract(const float* __restrict__ x, float* __restrict__ xyz, int BN) {
  int g = blockIdx.x * blockDim.x + threadIdx.x;
  if (g >= BN) return;
  const float* p = x + (size_t)g * 5;
  float* o = xyz + (size_t)g * 3;
  o[0] = p[0]; o[1] = p[1]; o[2] = p[2];
}

__global__ void k_input_mlp(const float* __restrict__ x,
                            const float* __restrict__ w1, const float* __restrict__ b1,
                            const float* __restrict__ w2, const float* __restrict__ b2,
                            float* __restrict__ out, int BN) {
  int g = blockIdx.x * blockDim.x + threadIdx.x;
  if (g >= BN) return;
  const float* xp = x + (size_t)g * 5;
  float h[32];
  for (int c = 0; c < 32; ++c) {
    float s = b1[c];
    for (int k = 0; k < 5; ++k) s += xp[k] * w1[k * 32 + c];
    h[c] = fmaxf(s, 0.f);
  }
  float* o = out + (size_t)g * 32;
  for (int c = 0; c < 32; ++c) {
    float s = b2[c];
    for (int k = 0; k < 32; ++k) s += h[k] * w2[k * 32 + c];
    o[c] = s;
  }
}

// ---------------- KNN (top-16 smallest sq-dist, low-index ties) ----------------
__global__ void k_knn(const float* __restrict__ qxyz, const float* __restrict__ rxyz,
                      int* __restrict__ idx, int B, int Nq, int Nr) {
  int g = blockIdx.x * blockDim.x + threadIdx.x;
  if (g >= B * Nq) return;
  int b = g / Nq;
  const float* qp = qxyz + (size_t)g * 3;
  float qx = qp[0], qy = qp[1], qz = qp[2];
  float bd[KNN]; int bi[KNN];
#pragma unroll
  for (int i = 0; i < KNN; ++i) { bd[i] = 3.4e38f; bi[i] = 0; }
  const float* rp = rxyz + (size_t)b * Nr * 3;
  for (int j = 0; j < Nr; ++j) {
    float dx = rp[j * 3] - qx, dy = rp[j * 3 + 1] - qy, dz = rp[j * 3 + 2] - qz;
    float d = dx * dx + dy * dy + dz * dz;
    if (d < bd[KNN - 1]) {
      int p = KNN - 1;
      while (p > 0 && bd[p - 1] > d) { bd[p] = bd[p - 1]; bi[p] = bi[p - 1]; --p; }
      bd[p] = d; bi[p] = j;
    }
  }
  int* o = idx + (size_t)g * KNN;
#pragma unroll
  for (int i = 0; i < KNN; ++i) o[i] = bi[i];
}

// ---------------- FPS (one workgroup per batch, serial over npoint) ----------------
__global__ __launch_bounds__(256) void k_fps(const float* __restrict__ xyz,
                                             int* __restrict__ out, int N, int np) {
  __shared__ float s_dist[4096];
  __shared__ float s_best[256];
  __shared__ int   s_bi[256];
  __shared__ int   s_far;
  const int b = blockIdx.x;
  const int t = threadIdx.x;
  const float* P = xyz + (size_t)b * N * 3;
  for (int i = t; i < N; i += 256) s_dist[i] = 1e10f;
  if (t == 0) s_far = 0;
  __syncthreads();
  for (int it = 0; it < np; ++it) {
    int far = s_far;
    if (t == 0) out[b * np + it] = far;
    float fx = P[far * 3], fy = P[far * 3 + 1], fz = P[far * 3 + 2];
    float best = -1.f; int bidx = 0;
    for (int i = t; i < N; i += 256) {
      float dx = P[i * 3] - fx, dy = P[i * 3 + 1] - fy, dz = P[i * 3 + 2] - fz;
      float d = dx * dx + dy * dy + dz * dz;
      float nd = fminf(s_dist[i], d);
      s_dist[i] = nd;
      if (nd > best) { best = nd; bidx = i; }
    }
    s_best[t] = best; s_bi[t] = bidx;
    __syncthreads();
    for (int s = 128; s > 0; s >>= 1) {
      if (t < s) {
        if (s_best[t + s] > s_best[t] ||
            (s_best[t + s] == s_best[t] && s_bi[t + s] < s_bi[t])) {
          s_best[t] = s_best[t + s]; s_bi[t] = s_bi[t + s];
        }
      }
      __syncthreads();
    }
    if (t == 0) s_far = s_bi[0];
    __syncthreads();
  }
}

// ---------------- WMMA GEMM with packed weights ----------------
// out[R,Cout] = A[R,Cin](f16,row-major) @ W(packed) ; 8 waves/block = 8 tiles.
// Total tile count (R/16 * Cout/16) must be a multiple of 8 (true for all calls).
__global__ __launch_bounds__(256) void k_gemm_wmma(
    const _Float16* __restrict__ A, const _Float16* __restrict__ Wp,
    const float* __restrict__ bias, const float* __restrict__ resid,
    float* __restrict__ outF, _Float16* __restrict__ outH,
    int R, int Cin, int Cout, int relu) {
  const int lane = threadIdx.x & 31;
  const int wv   = threadIdx.x >> 5;
  const int tile = blockIdx.x * 8 + wv;
  const int ctiles = Cout >> 4;
  const int tm = tile / ctiles;
  const int tn = tile - tm * ctiles;
  const int m  = lane & 15;
  const int kb = (lane >> 4) << 3;
  const int row = tm * 16 + m;
  const int col = tn * 16 + m;
  const int nks = Cin >> 5;
  const _Float16* bp = Wp + (((size_t)tn * nks) * 32 + lane) * 16;
  v8f acc = {};
  for (int ks = 0; ks < nks; ++ks) {
    v16h af;
    const _Float16* ap = A + (size_t)row * Cin + (ks << 5) + kb;
#pragma unroll
    for (int j = 0; j < 8; ++j) { af[j] = ap[j]; af[j + 8] = ap[16 + j]; }
    v16h bf = *(const v16h*)(bp + (size_t)ks * 512);
    if (ks + 1 < nks) __builtin_prefetch(bp + (size_t)(ks + 1) * 512, 0, 0);
    acc = __builtin_amdgcn_wmma_f32_16x16x32_f16(false, af, false, bf, (short)0, acc, false, false);
  }
  const int rbase = tm * 16 + ((lane >> 4) << 3);
  const float bb = bias ? bias[col] : 0.f;
#pragma unroll
  for (int r = 0; r < 8; ++r) {
    float v = acc[r] + bb;
    size_t o = (size_t)(rbase + r) * Cout + col;
    if (resid) v += resid[o];
    if (relu) v = fmaxf(v, 0.f);
    if (outF) outF[o] = v;
    if (outH) outH[o] = (_Float16)v;
  }
}

// ---------------- fused per-point attention ----------------
// 16x256 @ 256x256: A in LDS (row-major 16x256 f16), W packed in global.
// 8 waves, each covers 2 of the 16 column tiles.
template<bool OUT_H, bool RELU>
__device__ __forceinline__ void tile_gemm_256(
    const _Float16* Alds, const _Float16* __restrict__ Wp,
    const float* __restrict__ bias, float* outF, _Float16* outH) {
  const int lane = threadIdx.x & 31;
  const int wv = threadIdx.x >> 5;
  const int m = lane & 15;
  const int kb = (lane >> 4) << 3;
  const int rbase = (lane >> 4) << 3;
#pragma unroll
  for (int tt = 0; tt < 2; ++tt) {
    const int ct = wv + tt * 8;
    const int col = ct * 16 + m;
    const _Float16* bp = Wp + (((size_t)ct * 8) * 32 + lane) * 16;
    v8f acc = {};
#pragma unroll
    for (int ks = 0; ks < 8; ++ks) {
      v16h af;
      const _Float16* ap = Alds + m * DM + (ks << 5) + kb;
#pragma unroll
      for (int j = 0; j < 8; ++j) { af[j] = ap[j]; af[j + 8] = ap[16 + j]; }
      v16h bf = *(const v16h*)(bp + (size_t)ks * 512);
      acc = __builtin_amdgcn_wmma_f32_16x16x32_f16(false, af, false, bf, (short)0, acc, false, false);
    }
    const float bb = bias[col];
#pragma unroll
    for (int r = 0; r < 8; ++r) {
      float v = acc[r] + bb;
      if (RELU) v = fmaxf(v, 0.f);
      const int rr = rbase + r;
      if (OUT_H) outH[rr * DM + col] = (_Float16)v;
      else       outF[rr * DM + col] = v;
    }
  }
}

__global__ __launch_bounds__(256) void k_attn(
    const float* __restrict__ xyz, const int* __restrict__ knn,
    const float* __restrict__ q, const _Float16* __restrict__ kf,
    const _Float16* __restrict__ vf,
    const float* __restrict__ dw1, const float* __restrict__ db1,
    const _Float16* __restrict__ dw2, const float* __restrict__ db2,
    const _Float16* __restrict__ gw1, const float* __restrict__ gb1,
    const _Float16* __restrict__ gw2, const float* __restrict__ gb2,
    _Float16* __restrict__ res16, int B, int N) {
  __shared__ _Float16 s_h16[KNN * DM];  // pos1, then g1
  __shared__ float    s_pv [KNN * DM];  // pos, then v+pos
  __shared__ _Float16 s_a16[KNN * DM];  // a0 (f16)
  __shared__ float    s_af [KNN * DM];  // final a (f32)
  __shared__ float    s_q[DM];
  __shared__ float    s_rel[KNN * 3];
  __shared__ int      s_idx[KNN];

  const int bn = blockIdx.x;
  const int b = bn / N;
  const int t = threadIdx.x;

  if (t < KNN) {
    int id = knn[(size_t)bn * KNN + t];
    s_idx[t] = id;
    const float* pq = xyz + (size_t)bn * 3;
    const float* pk = xyz + ((size_t)b * N + id) * 3;
    s_rel[t * 3 + 0] = pq[0] - pk[0];
    s_rel[t * 3 + 1] = pq[1] - pk[1];
    s_rel[t * 3 + 2] = pq[2] - pk[2];
  }
  s_q[t] = q[(size_t)bn * DM + t];
  __syncthreads();

  { // pos1 = relu(rel @ dw1 + db1)  (3 -> 256, one column per thread)
    const int c = t;
    const float w0 = dw1[c], w1 = dw1[DM + c], w2 = dw1[2 * DM + c], bb = db1[c];
    for (int k = 0; k < KNN; ++k) {
      float v = s_rel[k * 3] * w0 + s_rel[k * 3 + 1] * w1 + s_rel[k * 3 + 2] * w2 + bb;
      s_h16[k * DM + c] = (_Float16)fmaxf(v, 0.f);
    }
  }
  __syncthreads();
  tile_gemm_256<false, false>(s_h16, dw2, db2, s_pv, nullptr);   // pos
  __syncthreads();
  { // a0 = q - k[n] + pos ; vp = v[n] + pos
    const int c = t;
    for (int k = 0; k < KNN; ++k) {
      size_t gi = ((size_t)b * N + s_idx[k]) * DM + c;
      float p = s_pv[k * DM + c];
      s_a16[k * DM + c] = (_Float16)(s_q[c] - (float)kf[gi] + p);
      s_pv[k * DM + c] = (float)vf[gi] + p;
    }
  }
  __syncthreads();
  tile_gemm_256<true, true >(s_a16, gw1, gb1, nullptr, s_h16);   // g1 = relu(a0@gw1+b)
  __syncthreads();
  tile_gemm_256<false, false>(s_h16, gw2, gb2, s_af, nullptr);   // a
  __syncthreads();
  { // channel-wise softmax over the K axis + weighted sum
    const int c = t;
    float mx = -1e30f;
    for (int k = 0; k < KNN; ++k) mx = fmaxf(mx, s_af[k * DM + c]);
    mx *= 0.0625f;
    float ssum = 0.f, racc = 0.f;
    for (int k = 0; k < KNN; ++k) {
      float e = __expf(s_af[k * DM + c] * 0.0625f - mx);
      ssum += e;
      racc += e * s_pv[k * DM + c];
    }
    res16[(size_t)bn * DM + c] = (_Float16)(racc / ssum);
  }
}

// ---------------- transition-down helpers ----------------
__global__ void k_newxyz(const float* __restrict__ xyz, const int* __restrict__ fidx,
                         float* __restrict__ nx, int B, int N, int np) {
  int g = blockIdx.x * blockDim.x + threadIdx.x;
  if (g >= B * np) return;
  int b = g / np;
  int id = fidx[g];
  const float* p = xyz + ((size_t)b * N + id) * 3;
  float* o = nx + (size_t)g * 3;
  o[0] = p[0]; o[1] = p[1]; o[2] = p[2];
}

__global__ void k_group(const float* __restrict__ xyz, const float* __restrict__ nxyz,
                        const float* __restrict__ pts, const int* __restrict__ idx,
                        float* __restrict__ h0, int B, int N, int np, int C) {
  int g = blockIdx.x * blockDim.x + threadIdx.x;
  if (g >= B * np * KNN) return;
  int k = g & (KNN - 1);
  int bn = g >> 4;
  int b = bn / np;
  int id = idx[(size_t)bn * KNN + k];
  const float* nx = nxyz + (size_t)bn * 3;
  const float* px = xyz + ((size_t)b * N + id) * 3;
  float* o = h0 + (size_t)g * (3 + C);
  o[0] = px[0] - nx[0]; o[1] = px[1] - nx[1]; o[2] = px[2] - nx[2];
  const float* pp = pts + ((size_t)b * N + id) * C;
  for (int c = 0; c < C; ++c) o[3 + c] = pp[c];
}

__global__ void k_linear(const float* __restrict__ A, const float* __restrict__ W,
                         const float* __restrict__ bias, float* __restrict__ out,
                         int R, int Cin, int Cout) {
  size_t g = (size_t)blockIdx.x * blockDim.x + threadIdx.x;
  if (g >= (size_t)R * Cout) return;
  int r = (int)(g / Cout), c = (int)(g % Cout);
  float s = bias[c];
  const float* a = A + (size_t)r * Cin;
  for (int k = 0; k < Cin; ++k) s += a[k] * W[(size_t)k * Cout + c];
  out[g] = s;
}

__global__ __launch_bounds__(256) void k_bn_stats(const float* __restrict__ z,
                                                  float* __restrict__ st, int R, int C) {
  __shared__ float ss[256], sq[256];
  int c = blockIdx.x, t = threadIdx.x;
  float s = 0.f, s2 = 0.f;
  for (int r = t; r < R; r += 256) {
    float v = z[(size_t)r * C + c];
    s += v; s2 += v * v;
  }
  ss[t] = s; sq[t] = s2;
  __syncthreads();
  for (int w = 128; w > 0; w >>= 1) {
    if (t < w) { ss[t] += ss[t + w]; sq[t] += sq[t + w]; }
    __syncthreads();
  }
  if (t == 0) {
    float m = ss[0] / R;
    st[2 * c] = m;
    st[2 * c + 1] = sq[0] / R - m * m;
  }
}

__global__ void k_bn_apply(const float* __restrict__ z, const float* __restrict__ st,
                           const float* __restrict__ g, const float* __restrict__ beta,
                           float* __restrict__ outF, _Float16* __restrict__ outH, int R, int C) {
  size_t i = (size_t)blockIdx.x * blockDim.x + threadIdx.x;
  if (i >= (size_t)R * C) return;
  int c = (int)(i % C);
  float m = st[2 * c], v = st[2 * c + 1];
  float y = fmaxf((z[i] - m) * rsqrtf(v + 1e-5f) * g[c] + beta[c], 0.f);
  if (outF) outF[i] = y;
  if (outH) outH[i] = (_Float16)y;
}

__global__ void k_maxpool(const float* __restrict__ h, float* __restrict__ out, int BN2, int C) {
  size_t g = (size_t)blockIdx.x * blockDim.x + threadIdx.x;
  if (g >= (size_t)BN2 * C) return;
  int r = (int)(g / C), c = (int)(g % C);
  const float* p = h + ((size_t)r * KNN) * C + c;
  float m = p[0];
  for (int k = 1; k < KNN; ++k) m = fmaxf(m, p[(size_t)k * C]);
  out[g] = m;
}

// ---------------- host orchestration ----------------
struct TBW {
  const float *fc1_b, *fc2_b, *dw1, *db1, *db2, *gb1, *gb2;
  const _Float16 *fc1_wp, *fc2_wp, *dw2_p, *gw1_p, *gw2_p, *wq_p, *wk_p, *wv_p;
  int C;
};
struct TDW {
  const float *w0, *b0, *b1, *g0, *g1, *be0, *be1;
  const _Float16 *w1_p;
};

extern "C" void kernel_launch(void* const* d_in, const int* in_sizes, int n_in,
                              void* d_out, int out_size, void* d_ws, size_t ws_size,
                              hipStream_t stream) {
  (void)in_sizes; (void)n_in; (void)out_size; (void)ws_size;
  const int B = 4;
  auto F = [&](int i) { return (const float*)d_in[i]; };

  // deterministic bump allocator over d_ws
  uintptr_t base = (uintptr_t)d_ws;
  size_t off = 0;
  auto alloc = [&](size_t bytes) -> void* {
    off = (off + 255) & ~(size_t)255;
    void* p = (void*)(base + off);
    off += bytes;
    return p;
  };
  // pack a weight for WMMA B-fragment consumption
  auto pack = [&](const float* src, int Cin, int Cout) -> const _Float16* {
    size_t n = (size_t)Cin * Cout;
    _Float16* d = (_Float16*)alloc(n * sizeof(_Float16));
    k_pack_w<<<(unsigned)((n + 255) / 256), 256, 0, stream>>>(src, d, Cin, Cout);
    return d;
  };

  // workspace buffers (sized for the worst stage; some are time-shared)
  float*    xyzA    = (float*)alloc((size_t)B * 4096 * 3 * 4);
  float*    xyzB    = (float*)alloc((size_t)B * 4096 * 3 * 4);
  float*    fA      = (float*)alloc((size_t)B * 4096 * 32 * 4);
  float*    fB      = (float*)alloc((size_t)B * 4096 * 32 * 4);
  _Float16* feats16 = (_Float16*)alloc((size_t)B * 4096 * 32 * 2);
  _Float16* x16     = (_Float16*)alloc((size_t)B * 4096 * 256 * 2); // also td h1(f16)
  float*    qz      = (float*)alloc((size_t)B * 4096 * 256 * 4);    // q / td z1
  float*    z2h0    = (float*)alloc((size_t)B * 4096 * 256 * 4);    // td h0 then z2
  _Float16* k16     = (_Float16*)alloc((size_t)B * 4096 * 256 * 2);
  _Float16* v16     = (_Float16*)alloc((size_t)B * 4096 * 256 * 2);
  _Float16* res16   = (_Float16*)alloc((size_t)B * 4096 * 256 * 2);
  int*      knnbuf  = (int*)alloc((size_t)B * 4096 * KNN * 4);
  int*      fpsi    = (int*)alloc((size_t)B * 1024 * 4);
  float*    stats   = (float*)alloc(512 * 2 * 4);

  // ---- parse inputs (setup_inputs() insertion order, recursively) ----
  int ii = 0;
  const float* X   = F(ii++);
  const float* wi1 = F(ii++); const float* bi1 = F(ii++);
  const float* wi2 = F(ii++); const float* bi2 = F(ii++);

  auto parse_tb = [&](int C) -> TBW {
    TBW t;
    const float* fc1_w = F(ii++); t.fc1_b = F(ii++);
    const float* fc2_w = F(ii++); t.fc2_b = F(ii++);
    t.dw1 = F(ii++); t.db1 = F(ii++);
    const float* dw2 = F(ii++); t.db2 = F(ii++);
    const float* gw1 = F(ii++); t.gb1 = F(ii++);
    const float* gw2 = F(ii++); t.gb2 = F(ii++);
    const float* wq = F(ii++); const float* wk = F(ii++); const float* wv = F(ii++);
    t.fc1_wp = pack(fc1_w, C, DM);
    t.fc2_wp = pack(fc2_w, DM, C);
    t.dw2_p  = pack(dw2, DM, DM);
    t.gw1_p  = pack(gw1, DM, DM);
    t.gw2_p  = pack(gw2, DM, DM);
    t.wq_p   = pack(wq, DM, DM);
    t.wk_p   = pack(wk, DM, DM);
    t.wv_p   = pack(wv, DM, DM);
    t.C = C;
    return t;
  };
  auto parse_td = [&](int ch) -> TDW {
    TDW t;
    t.w0 = F(ii++); const float* w1 = F(ii++);
    t.b0 = F(ii++); t.b1 = F(ii++);
    t.g0 = F(ii++); t.g1 = F(ii++);
    t.be0 = F(ii++); t.be1 = F(ii++);
    t.w1_p = pack(w1, ch, ch);
    return t;
  };

  TBW t0 = parse_tb(32);
  TDW td[4]; TBW tb[4];
  for (int i = 0; i < 4; ++i) {
    int ch = 64 << i;
    td[i] = parse_td(ch);
    tb[i] = parse_tb(ch);
  }

  auto gemm = [&](const _Float16* A, const _Float16* Wp, const float* bias,
                  const float* resid, float* outF, _Float16* outH,
                  int R, int Cin, int Cout, int relu) {
    unsigned tiles = (unsigned)((R / 16) * (Cout / 16));
    k_gemm_wmma<<<tiles / 8, 256, 0, stream>>>(A, Wp, bias, resid, outF, outH,
                                               R, Cin, Cout, relu);
  };

  auto run_tb = [&](const TBW& p, const float* xyz, const float* fin, float* fout, int N) {
    int BN = B * N;
    k_knn<<<(BN + 63) / 64, 64, 0, stream>>>(xyz, xyz, knnbuf, B, N, N);
    k_cvt_f16<<<(unsigned)(((size_t)BN * p.C + 255) / 256), 256, 0, stream>>>(fin, feats16, BN * p.C);
    gemm(feats16, p.fc1_wp, p.fc1_b, nullptr, nullptr, x16, BN, p.C, DM, 0);
    gemm(x16, p.wq_p, nullptr, nullptr, qz, nullptr, BN, DM, DM, 0);
    gemm(x16, p.wk_p, nullptr, nullptr, nullptr, k16, BN, DM, DM, 0);
    gemm(x16, p.wv_p, nullptr, nullptr, nullptr, v16, BN, DM, DM, 0);
    k_attn<<<BN, 256, 0, stream>>>(xyz, knnbuf, qz, k16, v16, p.dw1, p.db1, p.dw2_p, p.db2,
                                   p.gw1_p, p.gb1, p.gw2_p, p.gb2, res16, B, N);
    gemm(res16, p.fc2_wp, p.fc2_b, fin, fout, nullptr, BN, DM, p.C, 0);
  };

  auto run_td = [&](const TDW& p, const float* xyz, const float* fin, float* nxyz, float* fout,
                    int N, int np, int C, int ch) {
    k_fps<<<B, 256, 0, stream>>>(xyz, fpsi, N, np);
    int BQ = B * np;
    k_newxyz<<<(BQ + 63) / 64, 64, 0, stream>>>(xyz, fpsi, nxyz, B, N, np);
    k_knn<<<(BQ + 63) / 64, 64, 0, stream>>>(nxyz, xyz, knnbuf, B, np, N);
    int R = BQ * KNN;
    int inC = 3 + C;
    float* h0 = z2h0;
    k_group<<<(R + 63) / 64, 64, 0, stream>>>(xyz, nxyz, fin, knnbuf, h0, B, N, np, C);
    float* z1 = qz;
    k_linear<<<(unsigned)(((size_t)R * ch + 255) / 256), 256, 0, stream>>>(h0, p.w0, p.b0, z1, R, inC, ch);
    k_bn_stats<<<ch, 256, 0, stream>>>(z1, stats, R, ch);
    _Float16* h1 = x16;
    k_bn_apply<<<(unsigned)(((size_t)R * ch + 255) / 256), 256, 0, stream>>>(z1, stats, p.g0, p.be0, nullptr, h1, R, ch);
    float* z2 = z2h0; // h0 is dead now
    gemm(h1, p.w1_p, p.b1, nullptr, z2, nullptr, R, ch, ch, 0);
    k_bn_stats<<<ch, 256, 0, stream>>>(z2, stats, R, ch);
    k_bn_apply<<<(unsigned)(((size_t)R * ch + 255) / 256), 256, 0, stream>>>(z2, stats, p.g1, p.be1, z2, nullptr, R, ch);
    k_maxpool<<<(unsigned)(((size_t)BQ * ch + 255) / 256), 256, 0, stream>>>(z2, fout, BQ, ch);
  };

  // ---- pipeline ----
  int N = 4096;
  int BN = B * N;
  k_xyz_extract<<<(BN + 63) / 64, 64, 0, stream>>>(X, xyzA, BN);
  k_input_mlp<<<(BN + 63) / 64, 64, 0, stream>>>(X, wi1, bi1, wi2, bi2, fA, BN);

  float* xc = xyzA; float* xt = xyzB;
  float* fc = fA;   float* ft = fB;
  run_tb(t0, xc, fc, ft, N);
  { float* tmp = fc; fc = ft; ft = tmp; }
  int C = 32;
  for (int i = 0; i < 4; ++i) {
    int np = N / 4;
    int ch = 64 << i;
    run_td(td[i], xc, fc, xt, ft, N, np, C, ch);
    { float* tmp = xc; xc = xt; xt = tmp; }
    { float* tmp = fc; fc = ft; ft = tmp; }
    N = np; C = ch;
    float* out = (i == 3) ? (float*)d_out : ft;
    run_tb(tb[i], xc, fc, out, N);
    if (i != 3) { float* tmp = fc; fc = ft; ft = tmp; }
  }
}
